// BitConv2d_52587579572847
// MI455X (gfx1250) — matrix operand order repaired and checked by reading, not compiled, not run on hardware
//
#include <hip/hip_runtime.h>
#include <hip/hip_fp16.h>

typedef _Float16 v16h __attribute__((ext_vector_type(16)));
typedef _Float16 v8h  __attribute__((ext_vector_type(8)));
typedef float    v8f  __attribute__((ext_vector_type(8)));

typedef unsigned int u32x4 __attribute__((ext_vector_type(4)));
typedef int          i32x4 __attribute__((ext_vector_type(4)));
typedef int          i32x8 __attribute__((ext_vector_type(8)));

#define CIN   256
#define HH    56
#define WWID  56
#define PIX   (HH * WWID)        // 3136
#define NIMG  32
#define COUT  256
#define KTOT  (CIN * 9)          // 2304
#define MTOT  (NIMG * PIX)       // 100352
#define WELEM (COUT * CIN * 9)   // 589824
#define XELEM (NIMG * CIN * PIX) // 25690112
#define EPSQ  1e-6f

#if __has_builtin(__builtin_amdgcn_tensor_load_to_lds) && \
    __has_builtin(__builtin_amdgcn_s_wait_tensorcnt)
#define USE_TDM 1
#else
#define USE_TDM 0
#endif

// ---------------- reduction: mean|w| (deterministic two-stage) ----------------
__global__ void k_wsum_partial(const float* __restrict__ w, float* __restrict__ part) {
    __shared__ float red[256];
    float s = 0.0f;
    for (int i = blockIdx.x * 256 + threadIdx.x; i < WELEM; i += 256 * 256)
        s += fabsf(w[i]);
    red[threadIdx.x] = s;
    __syncthreads();
    for (int o = 128; o > 0; o >>= 1) {
        if ((int)threadIdx.x < o) red[threadIdx.x] += red[threadIdx.x + o];
        __syncthreads();
    }
    if (threadIdx.x == 0) part[blockIdx.x] = red[0];
}

__global__ void k_wsum_final(const float* __restrict__ part, float* __restrict__ wsum) {
    __shared__ float red[256];
    red[threadIdx.x] = part[threadIdx.x];
    __syncthreads();
    for (int o = 128; o > 0; o >>= 1) {
        if ((int)threadIdx.x < o) red[threadIdx.x] += red[threadIdx.x + o];
        __syncthreads();
    }
    if (threadIdx.x == 0) wsum[0] = fmaxf(red[0] * (1.0f / (float)WELEM), EPSQ);
}

// ---------------- per-channel activation amax ----------------
__global__ void k_amax(const float* __restrict__ x, float* __restrict__ amax) {
    const int ci = blockIdx.x;
    __shared__ float red[256];
    float m = 0.0f;
    for (int idx = threadIdx.x; idx < NIMG * PIX; idx += 256) {
        int n = idx / PIX, p = idx - n * PIX;
        m = fmaxf(m, fabsf(x[n * (CIN * PIX) + ci * PIX + p]));
    }
    red[threadIdx.x] = m;
    __syncthreads();
    for (int o = 128; o > 0; o >>= 1) {
        if ((int)threadIdx.x < o) red[threadIdx.x] = fmaxf(red[threadIdx.x], red[threadIdx.x + o]);
        __syncthreads();
    }
    if (threadIdx.x == 0) amax[ci] = fmaxf(red[0], EPSQ);
}

// ---------------- weight: ternary quant + fold per-channel act scale, fp16 [co][k] ----------------
__global__ void k_wxform(const float* __restrict__ w, const float* __restrict__ wsp,
                         const float* __restrict__ amax, _Float16* __restrict__ wf) {
    int i = blockIdx.x * 256 + threadIdx.x;   // WELEM is an exact multiple of 256
    float ws = wsp[0];
    int ci = (i / 9) % CIN;
    float q = fminf(fmaxf(rintf(w[i] * (1.0f / ws)), -1.0f), 1.0f);
    wf[i] = (_Float16)(q * ws * amax[ci] * (1.0f / 127.0f));
}

// ---------------- activation: int8 quant values stored exactly as fp16, NCHW ----------------
__global__ void k_axform(const float* __restrict__ x, const float* __restrict__ amax,
                         _Float16* __restrict__ af) {
    int i = blockIdx.x * 256 + threadIdx.x;   // XELEM is an exact multiple of 256
    int ci = (i / PIX) & (CIN - 1);
    float s = 127.0f / amax[ci];
    float q = fminf(fmaxf(rintf(x[i] * s), -128.0f), 127.0f);
    af[i] = (_Float16)q;
}

// ---------------- implicit-GEMM conv via v_wmma_f32_16x16x32_f16 + TDM B staging ----------------
// Block: 256 threads = 8 waves. Tile: BM=128, BN=128, BK=32.
// Wave (wid&3 -> M quarter, wid>>2 -> N half): 32x64 region = 2x4 WMMA tiles.
__global__ __launch_bounds__(256) void k_conv_wmma(
    const _Float16* __restrict__ Af,   // [N][Cin][H][W] fp16 (integer qx values)
    const _Float16* __restrict__ Wf,   // [Cout][Cin*9] fp16 (scale-folded)
    const float* __restrict__ bias,
    float* __restrict__ out) {         // [N][Cout][H][W] fp32

    __shared__ __align__(16) _Float16 As[128][40]; // [m][kk], 80B row stride (conflict-free)
    __shared__ __align__(16) _Float16 Bs[128][40]; // [n][kk]; TDM pads 64B rows to 80B stride
    __shared__ int   rBase[128];
    __shared__ short rH[128];
    __shared__ short rW[128];

    const int t  = threadIdx.x;
    const int m0 = blockIdx.x * 128;
    const int n0 = blockIdx.y * 128;

    if (t < 128) {
        int m = m0 + t;
        int n = m / PIX, p = m - n * PIX;
        rBase[t] = n * (CIN * PIX);
        rH[t] = (short)(p / WWID);
        rW[t] = (short)(p - (p / WWID) * WWID);
    }

    const int wid = t >> 5, lane = t & 31;
    const int hi = lane >> 4, lm = lane & 15;
    const int wm = (wid & 3) * 32;
    const int wn = (wid >> 2) * 64;

    v8f acc[2][4] = {};

    const int ai = t & 127;         // A row filled by this thread
    const int ak = (t >> 7) * 16;   // A kk base (0 or 16)
#if !USE_TDM
    const int bj = t >> 1;          // B col filled by this thread (fallback path)
    const int bk = (t & 1) * 16;    // B kk base (0 or 16)
#endif

#if USE_TDM
    // Uniform pieces of the Tensor DMA descriptor (per block).
    const unsigned long long wgBase =
        (unsigned long long)(const void*)(Wf + (size_t)n0 * KTOT); // byte address of tile col 0
    const unsigned ldsOff = (unsigned)(size_t)(&Bs[0][0]);         // LDS byte offset of Bs
    // Group1 word0: data_size=1(2B)<<16 | pad_enable<<20 | pad_interval=3<<22 (16 DW = 64B)
    //               | pad_amount=3<<25 (4 DW = 16B)  -> rows land at 80B stride.
    const int g1w0 = (1 << 16) | (1 << 20) | (3 << 22) | (3 << 25);
#endif

    for (int k0 = 0; k0 < KTOT; k0 += 32) {
        __syncthreads();

#if USE_TDM
        // ---- wave 0 kicks the TDM: 128x32-half tile of Wf -> Bs (padded rows) ----
        if (t < 32) {
            unsigned long long ga = wgBase + (unsigned)(k0 * 2);
            u32x4 g0;
            g0.x = 1u;                                        // count=1 valid descriptor
            g0.y = ldsOff;                                    // LDS destination
            g0.z = (unsigned)ga;                              // global_addr[31:0]
            g0.w = ((unsigned)(ga >> 32) & 0x01FFFFFFu) | 0x80000000u; // addr[56:32] | type=2
            i32x8 g1;
            g1[0] = g1w0;
            g1[1] = (int)((KTOT & 0xFFFF) << 16);             // tensor_dim0 = 2304 (lo16)
            g1[2] = (int)((COUT & 0xFFFF) << 16);             // tensor_dim1 = 256  (lo16)
            g1[3] = (int)(32 << 16);                          // tile_dim0 = 32 (K)
            g1[4] = 128;                                      // tile_dim1 = 128 (Cout rows)
            g1[5] = KTOT;                                     // tensor_dim0_stride = 2304
            g1[6] = 0;
            g1[7] = 0;
            i32x4 z4 = {};
#if __clang_major__ >= 23
            i32x8 z8 = {};
            __builtin_amdgcn_tensor_load_to_lds(g0, g1, z4, z4, z8, 0);
#else
            __builtin_amdgcn_tensor_load_to_lds(g0, g1, z4, z4, 0);
#endif
            __asm__ volatile("" ::: "memory");
        }
#else
        // ---- fallback: stage B tile with contiguous 32B vector loads ----
        {
            const _Float16* gp = Wf + (size_t)(n0 + bj) * KTOT + (k0 + bk);
            v8h w0 = *(const v8h*)(gp);
            v8h w1 = *(const v8h*)(gp + 8);
            *(v8h*)&Bs[bj][bk]     = w0;
            *(v8h*)&Bs[bj][bk + 8] = w1;
        }
#endif

        // ---- stage A tile: im2col with zero padding (overlaps the TDM transfer) ----
        {
            const int base = rBase[ai];
            const int h = rH[ai], w = rW[ai];
#pragma unroll
            for (int u = 0; u < 16; ++u) {
                int k  = k0 + ak + u;
                int ci = k / 9;
                int r  = k - ci * 9;
                int dh = r / 3;
                int dw = r - dh * 3;
                int hh = h + dh - 1;
                int ww = w + dw - 1;
                _Float16 v = (_Float16)0.0f;
                if ((unsigned)hh < (unsigned)HH && (unsigned)ww < (unsigned)WWID)
                    v = Af[base + ci * PIX + hh * WWID + ww];
                As[ai][ak + u] = v;
            }
            if (k0 + 32 < KTOT)  // prefetch next K-step's activation region
                __builtin_prefetch(&Af[base + ((k0 + 32 + ak) / 9) * PIX], 0, 0);
        }

#if USE_TDM
        if (t < 32) {
            __builtin_amdgcn_s_wait_tensorcnt(0);   // TDM done -> Bs fully written
            __asm__ volatile("" ::: "memory");
        }
#endif
        __syncthreads();

        // ---- build fragments per documented wave32 VGPR layouts ----
        v16h a[2], b[4];
#pragma unroll
        for (int mt = 0; mt < 2; ++mt) {
            const _Float16* ar = &As[wm + mt * 16 + lm][0];
            v8h a0 = *(const v8h*)(ar + 8 * hi);        // K 0..7   (+8 if hi)
            v8h a1 = *(const v8h*)(ar + 16 + 8 * hi);   // K 16..23 (+8 if hi)
            a[mt] = __builtin_shufflevector(a0, a1, 0, 1, 2, 3, 4, 5, 6, 7,
                                                    8, 9, 10, 11, 12, 13, 14, 15);
        }
#pragma unroll
        for (int nt = 0; nt < 4; ++nt) {
            const _Float16* br = &Bs[wn + nt * 16 + lm][8 * hi]; // column lm, K 0..7,16..23 (+8 if hi)
            v8h b0 = *(const v8h*)(br);
            v8h b1 = *(const v8h*)(br + 16);
            b[nt] = __builtin_shufflevector(b0, b1, 0, 1, 2, 3, 4, 5, 6, 7,
                                                    8, 9, 10, 11, 12, 13, 14, 15);
        }

#pragma unroll
        for (int mt = 0; mt < 2; ++mt)
#pragma unroll
            for (int nt = 0; nt < 4; ++nt)
                acc[mt][nt] = __builtin_amdgcn_wmma_f32_16x16x32_f16(
                    false, a[mt], false, b[nt], (short)0, acc[mt][nt], false, false);
    }

    // ---- epilogue: D layout = VGPR r -> row (r + 8*hi), lane%16 -> col ----
#pragma unroll
    for (int mt = 0; mt < 2; ++mt) {
#pragma unroll
        for (int nt = 0; nt < 4; ++nt) {
            const int col = n0 + wn + nt * 16 + lm;
            const float bv = bias[col];
#pragma unroll
            for (int r = 0; r < 8; ++r) {
                const int row  = wm + mt * 16 + hi * 8 + r;
                const int oidx = rBase[row] + col * PIX + rH[row] * WWID + rW[row];
                out[oidx] = acc[mt][nt][r] + bv;
            }
        }
    }
}

// ---------------- host side ----------------
extern "C" void kernel_launch(void* const* d_in, const int* in_sizes, int n_in,
                              void* d_out, int out_size, void* d_ws, size_t ws_size,
                              hipStream_t stream) {
    const float* x    = (const float*)d_in[0];
    const float* w    = (const float*)d_in[1];
    const float* bias = (const float*)d_in[2];
    float* out = (float*)d_out;

    // Workspace layout (needs ~52.6 MB):
    //   [0]      : ws scalar (clamped mean|w|)
    //   [256]    : 256 partial sums
    //   [2048]   : amax[256]
    //   [4096]   : wf16  (589824 * 2B)
    //   [+that]  : af16  (25690112 * 2B)
    char* wsb = (char*)d_ws;
    float*    wsum = (float*)(wsb + 0);
    float*    part = (float*)(wsb + 256);
    float*    amax = (float*)(wsb + 2048);
    _Float16* wf   = (_Float16*)(wsb + 4096);
    _Float16* af   = (_Float16*)(wsb + 4096 + (size_t)WELEM * 2);

    const size_t needed = 4096 + (size_t)WELEM * 2 + (size_t)XELEM * 2;
    if (ws_size < needed) return;  // cannot run without scratch

    k_wsum_partial<<<256, 256, 0, stream>>>(w, part);
    k_wsum_final  <<<1,   256, 0, stream>>>(part, wsum);
    k_amax        <<<256, 256, 0, stream>>>(x, amax);
    k_wxform      <<<WELEM / 256, 256, 0, stream>>>(w, wsum, amax, wf);
    k_axform      <<<XELEM / 256, 256, 0, stream>>>(x, amax, af);

    dim3 grid(MTOT / 128, COUT / 128);  // 784 x 2
    k_conv_wmma<<<grid, 256, 0, stream>>>(af, wf, bias, out);
}
